// GDNBlock_36910948942193
// MI455X (gfx1250) — matrix operand order repaired and checked by reading, not compiled, not run on hardware
//
#include <hip/hip_runtime.h>

// ---------------------------------------------------------------------------
// Gated DeltaNet block for MI455X (gfx1250, wave32, WMMA).
// bf16 WMMA (16x16x32, f32 accum) for all GEMMs + chunked delta-rule
// recurrence (chunk=64) so the sequential scan becomes WMMA matmuls.
// Async global->LDS staging (ASYNCcnt) for the serial chunk loop when the
// toolchain exposes the gfx1250 async builtins.
// ---------------------------------------------------------------------------

typedef __bf16 bf16;
typedef __attribute__((ext_vector_type(16))) __bf16 v16bf;
typedef __attribute__((ext_vector_type(8)))  __bf16 v8bf;
typedef __attribute__((ext_vector_type(8)))  float  v8f;

#define B_   2
#define T_   2048
#define D_   1024
#define H_   16
#define HD   64
#define KC   4
#define M_   (B_ * T_)          // 4096 rows
#define CHUNK 64
#define NCHUNK (T_ / CHUNK)     // 32 sequential chunk steps

#if defined(__gfx1250__) && __has_builtin(__builtin_amdgcn_global_load_async_to_lds_b128) && __has_builtin(__builtin_amdgcn_s_wait_asynccnt)
#define USE_ASYNC_LDS 1
#else
#define USE_ASYNC_LDS 0
#endif

union V16U { v16bf v; v8bf h[2]; };

// ---- WMMA fragment helpers (ISA 7.12.2 layouts, wave32) -------------------
// A-matrix 16x32 bf16: lanes 0-15 row M=lane, K = {0..7,16..23};
//                      lanes 16-31 row M=lane-16, K = {8..15,24..31}.
static __device__ inline v16bf fragA(const bf16* p, int ld) {
    int lane = threadIdx.x & 31;
    int r = lane & 15, hi = lane >> 4;
    const bf16* base = p + r * ld + hi * 8;
    V16U u;
    u.h[0] = *(const v8bf*)(base);        // K = hi*8 + 0..7
    u.h[1] = *(const v8bf*)(base + 16);   // K = 16 + hi*8 + 0..7
    return u.v;
}
// B-matrix 32x16 bf16: lane holds column n=lane&15, K = (lane<16?{0..7,16..23}:{8..15,24..31}).
// pT is the "transposed-stored" operand: pT[n*ld + k] = B[k][n], so each lane
// reads a contiguous row.
static __device__ inline v16bf fragB(const bf16* pT, int ld) {
    int lane = threadIdx.x & 31;
    int n = lane & 15, kh = (lane >> 4) * 8;
    const bf16* base = pT + n * ld + kh;
    V16U u;
    u.h[0] = *(const v8bf*)(base);
    u.h[1] = *(const v8bf*)(base + 16);
    return u.v;
}
static __device__ inline v8f wmma_bf16f32(v16bf a, v16bf b, v8f c) {
    return __builtin_amdgcn_wmma_f32_16x16x32_bf16(
        /*neg_a=*/false, a, /*neg_b=*/false, b,
        /*c_mod=*/(short)0, c, /*reuse_a=*/false, /*reuse_b=*/false);
}
// C/D 16x16 f32: lane holds column N=lane&15, rows M = (lane<16?0..7:8..15).
static __device__ inline void storeC(float* p, int ld, v8f c) {
    int lane = threadIdx.x & 31;
    int n = lane & 15, mo = (lane >> 4) * 8;
#pragma unroll
    for (int i = 0; i < 8; i++) p[(size_t)(mo + i) * ld + n] = c[i];
}

static __device__ inline v8f vzero8() {
    v8f z = {0.f, 0.f, 0.f, 0.f, 0.f, 0.f, 0.f, 0.f};
    return z;
}

#if USE_ASYNC_LDS
// The builtin's params (from the compiler diagnostic) are pointers to a
// 16-byte int vector: global source in addrspace(1), LDS dest in addrspace(3).
typedef int v4i __attribute__((vector_size(4 * sizeof(int))));
typedef __attribute__((address_space(1))) v4i* gv4p;
typedef __attribute__((address_space(3))) v4i* lv4p;
// one 16-byte async global->LDS packet
static __device__ inline void async_copy16(const void* g, void* l) {
    __builtin_amdgcn_global_load_async_to_lds_b128((gv4p)g, (lv4p)l, 0, 0);
}
#endif

// ---------------------------------------------------------------------------
// Elementwise converts
// ---------------------------------------------------------------------------
__global__ void cvt_x_kernel(const float* __restrict__ x, bf16* __restrict__ xb, int n) {
    int i = blockIdx.x * 256 + threadIdx.x;
    if (i < n) xb[i] = (bf16)x[i];
}

// Convert + transpose 5 weight matrices (DxD, f32 row-major [k][n]) into
// bf16 Wt[n][k] so WMMA B fragments are contiguous row loads.
__global__ void cvt_transpose_w(const float* __restrict__ W0, const float* __restrict__ W1,
                                const float* __restrict__ W2, const float* __restrict__ W3,
                                const float* __restrict__ W4, bf16* __restrict__ out) {
    const float* W;
    switch (blockIdx.z) {
        case 0: W = W0; break; case 1: W = W1; break;
        case 2: W = W2; break; case 3: W = W3; break;
        default: W = W4; break;
    }
    bf16* o = out + (size_t)blockIdx.z * D_ * D_;
    int kk = blockIdx.x * 16 + threadIdx.x;
    int n  = blockIdx.y * 16 + threadIdx.y;
    o[(size_t)n * D_ + kk] = (bf16)W[(size_t)kk * D_ + n];
}

// beta = sigmoid(x @ Wb), Wb (D,H)
__global__ void beta_kernel(const float* __restrict__ x, const float* __restrict__ Wb,
                            float* __restrict__ beta) {
    int i = blockIdx.x * 256 + threadIdx.x;   // i = m*H + h
    if (i >= M_ * H_) return;
    int m = i / H_, h = i % H_;
    float s = 0.f;
    for (int k = 0; k < D_; k++) s += x[(size_t)m * D_ + k] * Wb[k * H_ + h];
    beta[i] = 1.f / (1.f + __expf(-s));
}

// ---------------------------------------------------------------------------
// WMMA GEMM: Y[m][n] = sum_k A[m][k] * Wt[n][k].  M x N, K, bf16 in / f32 out.
// Block = 256 threads = 8 waves; block tile 128x128; wave tile 32x64.
// blockIdx.z selects one of up to 4 (W, Y) pairs (fused QKVG projection).
// ---------------------------------------------------------------------------
__global__ __launch_bounds__(256) void gemm_wmma(
    const bf16* __restrict__ A,
    const bf16* __restrict__ W0, const bf16* __restrict__ W1,
    const bf16* __restrict__ W2, const bf16* __restrict__ W3,
    float* __restrict__ Y0, float* __restrict__ Y1,
    float* __restrict__ Y2, float* __restrict__ Y3,
    int N, int K) {
    const bf16* Wt; float* Y;
    switch (blockIdx.z) {
        case 0: Wt = W0; Y = Y0; break;
        case 1: Wt = W1; Y = Y1; break;
        case 2: Wt = W2; Y = Y2; break;
        default: Wt = W3; Y = Y3; break;
    }
    int m0 = blockIdx.x * 128, n0 = blockIdx.y * 128;
    int w  = threadIdx.x >> 5;
    int m0w = m0 + (w & 3) * 32;   // 4 waves along M
    int n0w = n0 + (w >> 2) * 64;  // 2 waves along N

    v8f acc[2][4];
#pragma unroll
    for (int i = 0; i < 2; i++)
#pragma unroll
        for (int j = 0; j < 4; j++) acc[i][j] = vzero8();

    for (int k0 = 0; k0 < K; k0 += 32) {
        if (k0 + 64 < K) {  // pull next tiles toward L2/WGP$ (global_prefetch_b8)
            __builtin_prefetch(A + (size_t)m0w * K + k0 + 64, 0, 1);
            __builtin_prefetch(Wt + (size_t)n0w * K + k0 + 64, 0, 1);
        }
        v16bf a0 = fragA(A + (size_t)m0w * K + k0, K);
        v16bf a1 = fragA(A + (size_t)(m0w + 16) * K + k0, K);
#pragma unroll
        for (int j = 0; j < 4; j++) {
            v16bf b = fragB(Wt + (size_t)(n0w + j * 16) * K + k0, K);
            acc[0][j] = wmma_bf16f32(a0, b, acc[0][j]);
            acc[1][j] = wmma_bf16f32(a1, b, acc[1][j]);
        }
    }
#pragma unroll
    for (int i = 0; i < 2; i++)
#pragma unroll
        for (int j = 0; j < 4; j++)
            storeC(Y + (size_t)(m0w + i * 16) * N + n0w + j * 16, N, acc[i][j]);
}

// ---------------------------------------------------------------------------
// Depthwise causal conv (k=4) + SiLU (+ optional l2norm over head dim).
// One wave per (m=b*T+t, h); lane handles 2 channels.
// mode: 0 = q (l2norm * 1/sqrt(d)), 1 = k (l2norm), 2 = v (none).
// Output layout: ((b*H + h)*T + t)*HD + dd  (bf16, chunk-friendly).
// ---------------------------------------------------------------------------
__global__ void conv_silu_norm(const float* __restrict__ Y, const float* __restrict__ w,
                               const float* __restrict__ bias, bf16* __restrict__ out,
                               int mode) {
    int m = blockIdx.x, h = blockIdx.y, lane = threadIdx.x;
    int b = m / T_, t = m % T_;
    float acc[2];
#pragma unroll
    for (int e = 0; e < 2; e++) {
        int c = h * HD + lane * 2 + e;
        float a = bias[c];
#pragma unroll
        for (int j = 0; j < KC; j++) {
            int tt = t - (KC - 1) + j;
            if (tt >= 0) a += Y[((size_t)(b * T_ + tt)) * D_ + c] * w[c * KC + j];
        }
        acc[e] = a / (1.f + __expf(-a));  // SiLU
    }
    if (mode < 2) {
        float ss = acc[0] * acc[0] + acc[1] * acc[1];
#pragma unroll
        for (int off = 16; off >= 1; off >>= 1) ss += __shfl_xor(ss, off, 32);
        float rn = rsqrtf(ss + 1e-6f);
        if (mode == 0) rn *= 0.125f;  // 1/sqrt(64)
        acc[0] *= rn; acc[1] *= rn;
    }
    size_t base = (((size_t)(b * H_ + h)) * T_ + t) * HD + lane * 2;
    out[base]     = (bf16)acc[0];
    out[base + 1] = (bf16)acc[1];
}

// ---------------------------------------------------------------------------
// Chunked delta-rule recurrence.  One workgroup (256 thr / 8 waves) per (b,h).
// Per chunk (C=64, d=64):
//   G  = K K^T                         (WMMA)
//   A  = strict_lower(beta .* G)
//   R  = beta .* (V - K S0)            (K S0 via WMMA)
//   (I+A) E = R  -> forward substitution (f32, 64 serial rows)
//   O  = Q S0 + tril(Q K^T) E          (WMMA)
//   S  = S0 + K^T E                    (WMMA)
// LDS holds both orientations of every WMMA operand so all fragment loads
// are contiguous ds_load_b128.  Chunk staging uses async global->LDS copies
// (ASYNCcnt) where available.
// ---------------------------------------------------------------------------
__global__ __launch_bounds__(256) void delta_recurrence(
    const bf16* __restrict__ qg, const bf16* __restrict__ kg,
    const bf16* __restrict__ vg, const float* __restrict__ betag,
    float* __restrict__ og) {
    extern __shared__ char smem[];
    float* Sf  = (float*)smem;            // 64x64 f32 state S[k][v]
    float* Ef  = Sf + 64 * 64;            // holds K*S0 -> R -> E (in place)
    float* Gf  = Ef + 64 * 64;            // G -> A (in place)
    bf16*  Qs  = (bf16*)(Gf + 64 * 64);   // Q rows
    bf16*  Ks  = Qs + 64 * 64;            // K rows
    bf16*  KTs = Ks + 64 * 64;            // K^T rows: KTs[k][c] = K[c][k]
    bf16*  STb = KTs + 64 * 64;           // S^T bf16: STb[v][k] = S[k][v]
    bf16*  ETb = STb + 64 * 64;           // E^T bf16: ETb[v][c] = E[c][v]
    bf16*  Mb  = ETb + 64 * 64;           // tril(Q K^T) rows, bf16
    bf16*  Vs  = Mb + 64 * 64;            // V rows
    float* beta_s = (float*)(Vs + 64 * 64);

    int bh = blockIdx.x;
    int b = bh / H_, h = bh % H_;
    const size_t hb = (size_t)bh * T_ * HD;
    int tid = threadIdx.x;
    int wv  = tid >> 5;

    for (int i = tid; i < 64 * 64; i += 256) { Sf[i] = 0.f; STb[i] = (bf16)0.f; }
    __syncthreads();

    for (int ch = 0; ch < NCHUNK; ch++) {
        int t0 = ch * CHUNK;
        const size_t cb = hb + (size_t)t0 * HD;   // chunk base (contiguous 64*64)
#if USE_ASYNC_LDS
        // ---- async global->LDS staging of Q,K,V chunk ----
        for (int i = tid; i < 512; i += 256) {    // 512 x 16B packets / matrix
            int off = i * 8;                      // element offset (8 bf16 = 16B)
            async_copy16(qg + cb + off, Qs + off);
            async_copy16(kg + cb + off, Ks + off);
            async_copy16(vg + cb + off, Vs + off);
        }
        if (tid < 64) beta_s[tid] = betag[((size_t)(b * T_ + t0 + tid)) * H_ + h];
        __builtin_amdgcn_s_wait_asynccnt(0);
        __syncthreads();
        // build K^T in LDS
        for (int i = tid; i < 64 * 64; i += 256) {
            int r = i >> 6, cc = i & 63;
            KTs[cc * 64 + r] = Ks[i];
        }
        __syncthreads();
#else
        // ---- synchronous chunk load fallback ----
        for (int i = tid; i < 64 * 64; i += 256) {
            int r = i >> 6, cc = i & 63;
            bf16 qv = qg[cb + i];
            bf16 kv = kg[cb + i];
            Qs[i] = qv; Ks[i] = kv; Vs[i] = vg[cb + i];
            KTs[cc * 64 + r] = kv;
        }
        if (tid < 64) beta_s[tid] = betag[((size_t)(b * T_ + t0 + tid)) * H_ + h];
        __syncthreads();
#endif

        // ---- G = K K^T -> Gf ; KS0 = K @ S0 -> Ef ----
#pragma unroll
        for (int tt = 0; tt < 2; tt++) {
            int tile = wv * 2 + tt, ti = tile >> 2, tj = tile & 3;
            v8f accG = vzero8(), accR = vzero8();
#pragma unroll
            for (int ks = 0; ks < 64; ks += 32) {
                v16bf a = fragA(Ks + ti * 16 * 64 + ks, 64);
                accG = wmma_bf16f32(a, fragB(Ks  + tj * 16 * 64 + ks, 64), accG);
                accR = wmma_bf16f32(a, fragB(STb + tj * 16 * 64 + ks, 64), accR);
            }
            storeC(Gf + ti * 16 * 64 + tj * 16, 64, accG);
            storeC(Ef + ti * 16 * 64 + tj * 16, 64, accR);
        }
        __syncthreads();

        // ---- R = beta*(V - KS0) (into Ef) ; A = strict-lower beta*G ----
        for (int i = tid; i < 64 * 64; i += 256) {
            int r = i >> 6, cc = i & 63;
            float vv = (float)Vs[i];
            Ef[i] = beta_s[r] * (vv - Ef[i]);
            Gf[i] = (cc < r) ? beta_s[r] * Gf[i] : 0.f;
        }
        __syncthreads();

        // ---- forward substitution: E_i = R_i - sum_{j<i} A_ij E_j (in place) ----
        for (int i = 0; i < 64; i++) {
            if (tid < 64) {
                float s = 0.f;
                for (int j = 0; j < i; j++) s += Gf[i * 64 + j] * Ef[j * 64 + tid];
                Ef[i * 64 + tid] -= s;
            }
            __syncthreads();
        }
        // E^T bf16
        for (int i = tid; i < 64 * 64; i += 256) {
            int r = i >> 6, cc = i & 63;           // i = v*64 + c
            ETb[i] = (bf16)Ef[cc * 64 + r];
        }
        __syncthreads();

        // ---- M = tril(Q K^T) (incl diag), masked bf16 store ----
#pragma unroll
        for (int tt = 0; tt < 2; tt++) {
            int tile = wv * 2 + tt, ti = tile >> 2, tj = tile & 3;
            v8f acc = vzero8();
#pragma unroll
            for (int ks = 0; ks < 64; ks += 32)
                acc = wmma_bf16f32(fragA(Qs + ti * 16 * 64 + ks, 64),
                                   fragB(Ks + tj * 16 * 64 + ks, 64), acc);
            int lane = tid & 31, n = lane & 15, mo = (lane >> 4) * 8;
#pragma unroll
            for (int ii = 0; ii < 8; ii++) {
                int I = ti * 16 + mo + ii, J = tj * 16 + n;
                Mb[I * 64 + J] = (J <= I) ? (bf16)acc[ii] : (bf16)0.f;
            }
        }
        __syncthreads();

        // ---- O = Q S0 + M E -> global ; S += K^T E ----
#pragma unroll
        for (int tt = 0; tt < 2; tt++) {
            int tile = wv * 2 + tt, ti = tile >> 2, tj = tile & 3;
            v8f accO = vzero8(), accS = vzero8();
#pragma unroll
            for (int ks = 0; ks < 64; ks += 32) {
                accO = wmma_bf16f32(fragA(Qs  + ti * 16 * 64 + ks, 64),
                                    fragB(STb + tj * 16 * 64 + ks, 64), accO);
                accO = wmma_bf16f32(fragA(Mb  + ti * 16 * 64 + ks, 64),
                                    fragB(ETb + tj * 16 * 64 + ks, 64), accO);
                accS = wmma_bf16f32(fragA(KTs + ti * 16 * 64 + ks, 64),
                                    fragB(ETb + tj * 16 * 64 + ks, 64), accS);
            }
            int lane = tid & 31, n = lane & 15, mo = (lane >> 4) * 8;
#pragma unroll
            for (int ii = 0; ii < 8; ii++) {
                int t = t0 + ti * 16 + mo + ii, dd = tj * 16 + n;
                og[(((size_t)(b * T_ + t)) * H_ + h) * HD + dd] = accO[ii];
                Sf[(ti * 16 + mo + ii) * 64 + tj * 16 + n] += accS[ii];
            }
        }
        __syncthreads();

        // ---- refresh S^T bf16 for next chunk ----
        for (int i = tid; i < 64 * 64; i += 256) {
            int r = i >> 6, cc = i & 63;           // i = v*64 + k
            STb[i] = (bf16)Sf[cc * 64 + r];
        }
        __syncthreads();
    }
}

// ---------------------------------------------------------------------------
// RMSNorm over head dim + SiLU gate; pack bf16 rows for final GEMM.
// ---------------------------------------------------------------------------
__global__ void out_norm_gate(const float* __restrict__ og, const float* __restrict__ Yg,
                              const float* __restrict__ nw, bf16* __restrict__ ob) {
    int m = blockIdx.x, h = blockIdx.y, lane = threadIdx.x;
    size_t base = ((size_t)m * H_ + h) * HD + lane * 2;
    float o0 = og[base], o1 = og[base + 1];
    float ss = o0 * o0 + o1 * o1;
#pragma unroll
    for (int off = 16; off >= 1; off >>= 1) ss += __shfl_xor(ss, off, 32);
    float rms = rsqrtf(ss * (1.f / HD) + 1e-5f);
    int dd = lane * 2;
    float g0 = Yg[(size_t)m * D_ + h * HD + dd];
    float g1 = Yg[(size_t)m * D_ + h * HD + dd + 1];
    g0 = g0 / (1.f + __expf(-g0));
    g1 = g1 / (1.f + __expf(-g1));
    ob[(size_t)m * D_ + h * HD + dd]     = (bf16)(o0 * rms * nw[dd] * g0);
    ob[(size_t)m * D_ + h * HD + dd + 1] = (bf16)(o1 * rms * nw[dd + 1] * g1);
}

// ---------------------------------------------------------------------------
extern "C" void kernel_launch(void* const* d_in, const int* in_sizes, int n_in,
                              void* d_out, int out_size, void* d_ws, size_t ws_size,
                              hipStream_t stream) {
    (void)in_sizes; (void)n_in; (void)out_size; (void)ws_size;
    const float* x   = (const float*)d_in[0];
    const float* Wq  = (const float*)d_in[1];
    const float* Wk  = (const float*)d_in[2];
    const float* Wv  = (const float*)d_in[3];
    const float* Wo  = (const float*)d_in[4];
    const float* Wg  = (const float*)d_in[5];
    const float* Wb  = (const float*)d_in[6];
    const float* cqw = (const float*)d_in[7];
    const float* cqb = (const float*)d_in[8];
    const float* ckw = (const float*)d_in[9];
    const float* ckb = (const float*)d_in[10];
    const float* cvw = (const float*)d_in[11];
    const float* cvb = (const float*)d_in[12];
    const float* nw  = (const float*)d_in[13];
    float* out = (float*)d_out;

    char* ws = (char*)d_ws;
    size_t off = 0;
    auto alloc = [&](size_t bytes) -> void* {
        void* p = ws + off;
        off = (off + bytes + 255) & ~(size_t)255;
        return p;
    };
    const size_t DD = (size_t)D_ * D_;
    bf16*  xb   = (bf16*)alloc((size_t)M_ * D_ * 2);
    bf16*  Wt   = (bf16*)alloc(5 * DD * 2);      // [Wq,Wk,Wv,Wg,Wo]^T bf16
    float* Yq   = (float*)alloc((size_t)M_ * D_ * 4);
    float* Yk   = (float*)alloc((size_t)M_ * D_ * 4);
    float* Yv   = (float*)alloc((size_t)M_ * D_ * 4);
    float* Yg   = (float*)alloc((size_t)M_ * D_ * 4);
    bf16*  qb   = (bf16*)alloc((size_t)M_ * D_ * 2);
    bf16*  kb   = (bf16*)alloc((size_t)M_ * D_ * 2);
    bf16*  vb   = (bf16*)alloc((size_t)M_ * D_ * 2);
    float* beta = (float*)alloc((size_t)M_ * H_ * 4);
    float* og   = (float*)alloc((size_t)M_ * D_ * 4);
    bf16*  ob   = (bf16*)alloc((size_t)M_ * D_ * 2);

    // 1) precision conversion + weight transpose
    cvt_x_kernel<<<(M_ * D_ + 255) / 256, 256, 0, stream>>>(x, xb, M_ * D_);
    cvt_transpose_w<<<dim3(D_ / 16, D_ / 16, 5), dim3(16, 16), 0, stream>>>(
        Wq, Wk, Wv, Wg, Wo, Wt);
    beta_kernel<<<(M_ * H_ + 255) / 256, 256, 0, stream>>>(x, Wb, beta);

    // 2) fused QKVG projections (WMMA)
    gemm_wmma<<<dim3(M_ / 128, D_ / 128, 4), 256, 0, stream>>>(
        xb, Wt, Wt + DD, Wt + 2 * DD, Wt + 3 * DD, Yq, Yk, Yv, Yg, D_, D_);

    // 3) causal conv + SiLU (+ l2norm for q,k)
    conv_silu_norm<<<dim3(M_, H_), 32, 0, stream>>>(Yq, cqw, cqb, qb, 0);
    conv_silu_norm<<<dim3(M_, H_), 32, 0, stream>>>(Yk, ckw, ckb, kb, 1);
    conv_silu_norm<<<dim3(M_, H_), 32, 0, stream>>>(Yv, cvw, cvb, vb, 2);

    // 4) chunked delta-rule recurrence (WMMA in LDS, async chunk staging)
    size_t smem = 3 * 64 * 64 * 4      // Sf, Ef, Gf (f32)
                + 7 * 64 * 64 * 2      // Qs, Ks, KTs, STb, ETb, Mb, Vs (bf16)
                + 64 * 4 + 256;        // beta + pad
    delta_recurrence<<<B_ * H_, 256, smem, stream>>>(qb, kb, vb, beta, og);

    // 5) RMSNorm + gate, then output projection (WMMA)
    out_norm_gate<<<dim3(M_, H_), 32, 0, stream>>>(og, Yg, nw, ob);
    gemm_wmma<<<dim3(M_ / 128, D_ / 128, 1), 256, 0, stream>>>(
        ob, Wt + 4 * DD, Wt + 4 * DD, Wt + 4 * DD, Wt + 4 * DD,
        out, out, out, out, D_, D_);
}
